// WindowAttention3D_39170101739820
// MI455X (gfx1250) — matrix operand order repaired and checked by reading, not compiled, not run on hardware
//
#include <hip/hip_runtime.h>

#define NTOK 64      // tokens per window (4^3)
#define DIM  128
#define NHEAD 4
#define HD   32
#define NWIN 392
#define BWIN 3136    // NW * B

typedef __attribute__((ext_vector_type(16))) _Float16 v16h;
typedef __attribute__((ext_vector_type(8)))  _Float16 v8h;
typedef __attribute__((ext_vector_type(8)))  float    v8f;

// LDS row strides (halves), padded +8 to break bank conflicts on row-strided loads
#define XS_STRIDE 136   // [64][128] x / attention-out (reused)
#define QK_STRIDE 40    // [H][64][32] q, k
#define VT_STRIDE 72    // [H][32][64] v transposed
#define P_STRIDE  72    // [H][64][64] softmax probs (f16)

__device__ __forceinline__ v16h make_frag(const _Float16* p0, const _Float16* p1) {
  v8h lo = *(const v8h*)p0;
  v8h hi = *(const v8h*)p1;
  return __builtin_shufflevector(lo, hi, 0,1,2,3,4,5,6,7,8,9,10,11,12,13,14,15);
}

__device__ __forceinline__ v8f wmma_f16(v16h a, v16h b, v8f c) {
  return __builtin_amdgcn_wmma_f32_16x16x32_f16(false, a, false, b, (short)0, c, false, false);
}

__device__ __forceinline__ void store_pk(_Float16* p, float a, float b) {
  union { _Float16 h2[2]; unsigned u; } pk;
  pk.h2[0] = (_Float16)a;
  pk.h2[1] = (_Float16)b;
  *(unsigned*)p = pk.u;
}

// ---------------------------------------------------------------------------
// Prep: f32 weights -> f16, and gather rpb_table[rel_index] -> dense [H][N][N]
// ---------------------------------------------------------------------------
__global__ void prep_kernel(const float* __restrict__ qkv_w,
                            const float* __restrict__ proj_w,
                            const float* __restrict__ tbl,
                            const int*   __restrict__ rel_index,
                            _Float16* __restrict__ qkv_wh,
                            _Float16* __restrict__ proj_wh,
                            float*    __restrict__ rpe) {
  int i = blockIdx.x * blockDim.x + threadIdx.x;
  if (i < 3 * DIM * DIM) qkv_wh[i] = (_Float16)qkv_w[i];
  if (i < DIM * DIM)     proj_wh[i] = (_Float16)proj_w[i];
  if (i < NHEAD * NTOK * NTOK) {
    int h  = i / (NTOK * NTOK);
    int mn = i % (NTOK * NTOK);
    rpe[i] = tbl[rel_index[mn] * NHEAD + h];
  }
}

// ---------------------------------------------------------------------------
// Fused per-window attention: one workgroup (8 wave32s) per window
// ---------------------------------------------------------------------------
__global__ void __launch_bounds__(256) win_attn_kernel(
    const float* __restrict__ x, const float* __restrict__ mask,
    const float* __restrict__ qkv_b, const float* __restrict__ proj_b,
    const _Float16* __restrict__ qkv_wh, const _Float16* __restrict__ proj_wh,
    const float* __restrict__ rpe, float* __restrict__ out) {

  __shared__ __align__(16) _Float16 sh_x[NTOK * XS_STRIDE];             // x, then attn-out
  __shared__ __align__(16) _Float16 sh_q[NHEAD][NTOK][QK_STRIDE];
  __shared__ __align__(16) _Float16 sh_k[NHEAD][NTOK][QK_STRIDE];
  __shared__ __align__(16) _Float16 sh_vT[NHEAD][HD][VT_STRIDE];
  __shared__ __align__(16) _Float16 sh_P[NHEAD][NTOK][P_STRIDE];

  const int b     = blockIdx.x;
  const int tid   = threadIdx.x;
  const int wvu   = __builtin_amdgcn_readfirstlane(tid >> 5);  // wave id in SGPR
  const int lane  = tid & 31;
  const int lhalf = lane & 15;
  const int lhi   = lane >> 4;

  // ---- stage 1: load x tile (64x128 f32) -> f16 LDS --------------------
  const float4* xg = (const float4*)(x + (size_t)b * NTOK * DIM);
  for (int i = tid; i < NTOK * DIM / 4; i += 256) {
    float4 v = xg[i];
    int row = (i * 4) / DIM, col = (i * 4) % DIM;
    _Float16* d = &sh_x[row * XS_STRIDE + col];
    d[0] = (_Float16)v.x; d[1] = (_Float16)v.y;
    d[2] = (_Float16)v.z; d[3] = (_Float16)v.w;
  }
  __syncthreads();

  // ---- stage 2: QKV GEMM [64,128]@[128,384]^T, 96 tiles / 8 waves ------
  const float scale = 0.1767766952966369f;   // 1/sqrt(32)
  for (int ntl = 0; ntl < 3; ++ntl) {
    const int jt = wvu * 3 + ntl;        // tile column index 0..23 (scalar)
    const int j0 = jt * 16;              // scalar
    const int which = j0 >> 7;           // scalar: 0=q 1=k 2=v
    const int h  = (j0 >> 5) & 3;        // scalar head
    const int d0 = j0 & 31;              // scalar: 0 or 16
    const int j  = j0 + lhalf;           // per-lane output column (B rows, bias)
    const _Float16* bbase = qkv_wh + j * DIM;
    const float bias = qkv_b[j];

    // preload the 4 B fragments for this output-column tile (K = 128)
    v16h bf0 = make_frag(bbase +   0 + lhi * 8, bbase +   0 + lhi * 8 + 16);
    v16h bf1 = make_frag(bbase +  32 + lhi * 8, bbase +  32 + lhi * 8 + 16);
    v16h bf2 = make_frag(bbase +  64 + lhi * 8, bbase +  64 + lhi * 8 + 16);
    v16h bf3 = make_frag(bbase +  96 + lhi * 8, bbase +  96 + lhi * 8 + 16);

    v8f acc[4];
    #pragma unroll
    for (int mt = 0; mt < 4; ++mt) {
      const _Float16* arow = &sh_x[(mt * 16 + lhalf) * XS_STRIDE + lhi * 8];
      v8f a = {};
      a = wmma_f16(make_frag(arow +  0, arow + 16), bf0, a);
      a = wmma_f16(make_frag(arow + 32, arow + 48), bf1, a);
      a = wmma_f16(make_frag(arow + 64, arow + 80), bf2, a);
      a = wmma_f16(make_frag(arow + 96, arow + 112), bf3, a);
      acc[mt] = a;
    }

    // Uniform (scalar) branch on q/k/v: constant LDS strides inside each arm
    if (which == 0) {
      _Float16* dst = &sh_q[h][lhi * 8][d0 + lhalf];
      #pragma unroll
      for (int mt = 0; mt < 4; ++mt)
        #pragma unroll
        for (int r = 0; r < 8; ++r)
          dst[(mt * 16 + r) * QK_STRIDE] = (_Float16)((acc[mt][r] + bias) * scale);
    } else if (which == 1) {
      _Float16* dst = &sh_k[h][lhi * 8][d0 + lhalf];
      #pragma unroll
      for (int mt = 0; mt < 4; ++mt)
        #pragma unroll
        for (int r = 0; r < 8; ++r)
          dst[(mt * 16 + r) * QK_STRIDE] = (_Float16)(acc[mt][r] + bias);
    } else {
      _Float16* dst = &sh_vT[h][d0 + lhalf][lhi * 8];   // contiguous along m: pack pairs
      #pragma unroll
      for (int mt = 0; mt < 4; ++mt)
        #pragma unroll
        for (int r = 0; r < 8; r += 2)
          store_pk(dst + mt * 16 + r, acc[mt][r] + bias, acc[mt][r + 1] + bias);
    }
  }
  __syncthreads();

  // ---- stage 3: per-head attention (2 waves per head) -------------------
  {
    const int h   = wvu >> 1;                     // scalar
    const int mtb = (wvu & 1) * 2;                // scalar: rows mtb*16 .. mtb*16+31
    const float* rpeH  = rpe  + h * NTOK * NTOK;
    const float* maskW = mask + (size_t)(b % NWIN) * NTOK * NTOK;

    v8f S[2][4];
    #pragma unroll
    for (int mt2 = 0; mt2 < 2; ++mt2) {
      const int m0 = (mtb + mt2) * 16;
      const _Float16* ap = &sh_q[h][m0 + lhalf][lhi * 8];
      v16h a = make_frag(ap, ap + 16);
      #pragma unroll
      for (int nt = 0; nt < 4; ++nt) {
        const _Float16* bp = &sh_k[h][nt * 16 + lhalf][lhi * 8];
        v16h bf = make_frag(bp, bp + 16);
        v8f acc = {};
        acc = wmma_f16(a, bf, acc);
        const int n = nt * 16 + lhalf;
        #pragma unroll
        for (int r = 0; r < 8; ++r) {
          const int m = m0 + r + lhi * 8;
          acc[r] += rpeH[m * NTOK + n] + maskW[m * NTOK + n];
        }
        S[mt2][nt] = acc;
      }
    }

    // softmax fully in registers: each row lives in one 16-lane half
    #pragma unroll
    for (int mt2 = 0; mt2 < 2; ++mt2) {
      #pragma unroll
      for (int r = 0; r < 8; ++r) {
        float mx = fmaxf(fmaxf(S[mt2][0][r], S[mt2][1][r]),
                         fmaxf(S[mt2][2][r], S[mt2][3][r]));
        mx = fmaxf(mx, __shfl_xor(mx, 1));
        mx = fmaxf(mx, __shfl_xor(mx, 2));
        mx = fmaxf(mx, __shfl_xor(mx, 4));
        mx = fmaxf(mx, __shfl_xor(mx, 8));
        float sum = 0.f;
        #pragma unroll
        for (int nt = 0; nt < 4; ++nt) {
          float e = __expf(S[mt2][nt][r] - mx);
          S[mt2][nt][r] = e;
          sum += e;
        }
        sum += __shfl_xor(sum, 1);
        sum += __shfl_xor(sum, 2);
        sum += __shfl_xor(sum, 4);
        sum += __shfl_xor(sum, 8);
        const float inv = 1.f / sum;
        const int m = (mtb + mt2) * 16 + r + lhi * 8;
        #pragma unroll
        for (int nt = 0; nt < 4; ++nt)
          sh_P[h][m][nt * 16 + lhalf] = (_Float16)(S[mt2][nt][r] * inv);
      }
    }
    __syncthreads();

    // P @ V  ->  attention output into sh_x (reused) at columns h*32..h*32+31
    #pragma unroll
    for (int mt2 = 0; mt2 < 2; ++mt2) {
      const int m0 = (mtb + mt2) * 16;
      #pragma unroll
      for (int dt = 0; dt < 2; ++dt) {
        v8f acc = {};
        #pragma unroll
        for (int ks = 0; ks < 2; ++ks) {
          const _Float16* ap = &sh_P[h][m0 + lhalf][ks * 32 + lhi * 8];
          v16h a = make_frag(ap, ap + 16);
          const _Float16* bp = &sh_vT[h][dt * 16 + lhalf][ks * 32 + lhi * 8];
          v16h bf = make_frag(bp, bp + 16);
          acc = wmma_f16(a, bf, acc);
        }
        _Float16* dst = &sh_x[(m0 + lhi * 8) * XS_STRIDE + h * 32 + dt * 16 + lhalf];
        #pragma unroll
        for (int r = 0; r < 8; ++r)
          dst[r * XS_STRIDE] = (_Float16)acc[r];
      }
    }
  }
  __syncthreads();

  // ---- stage 4: output projection [64,128]@[128,128]^T + bias ----------
  {
    const int j = wvu * 16 + lhalf;
    const _Float16* bbase = proj_wh + j * DIM;
    const float bias = proj_b[j];
    float* outp = out + (size_t)b * NTOK * DIM;

    v16h bf0 = make_frag(bbase +   0 + lhi * 8, bbase +   0 + lhi * 8 + 16);
    v16h bf1 = make_frag(bbase +  32 + lhi * 8, bbase +  32 + lhi * 8 + 16);
    v16h bf2 = make_frag(bbase +  64 + lhi * 8, bbase +  64 + lhi * 8 + 16);
    v16h bf3 = make_frag(bbase +  96 + lhi * 8, bbase +  96 + lhi * 8 + 16);

    #pragma unroll
    for (int mt = 0; mt < 4; ++mt) {
      const _Float16* arow = &sh_x[(mt * 16 + lhalf) * XS_STRIDE + lhi * 8];
      v8f acc = {};
      acc = wmma_f16(make_frag(arow +  0, arow + 16), bf0, acc);
      acc = wmma_f16(make_frag(arow + 32, arow + 48), bf1, acc);
      acc = wmma_f16(make_frag(arow + 64, arow + 80), bf2, acc);
      acc = wmma_f16(make_frag(arow + 96, arow + 112), bf3, acc);
      float* op = outp + (mt * 16 + lhi * 8) * DIM + j;
      #pragma unroll
      for (int r = 0; r < 8; ++r)
        op[r * DIM] = acc[r] + bias;
    }
  }
}

// ---------------------------------------------------------------------------
extern "C" void kernel_launch(void* const* d_in, const int* in_sizes, int n_in,
                              void* d_out, int out_size, void* d_ws, size_t ws_size,
                              hipStream_t stream) {
  (void)in_sizes; (void)n_in; (void)out_size; (void)ws_size;
  const float* x      = (const float*)d_in[0];
  const float* mask   = (const float*)d_in[1];
  const float* qkv_w  = (const float*)d_in[2];
  const float* qkv_b  = (const float*)d_in[3];
  const float* proj_w = (const float*)d_in[4];
  const float* proj_b = (const float*)d_in[5];
  const float* tbl    = (const float*)d_in[6];
  const int*   relidx = (const int*)d_in[7];

  char* ws = (char*)d_ws;
  _Float16* qkv_wh  = (_Float16*)ws;                       // 384*128*2   = 98304 B
  _Float16* proj_wh = (_Float16*)(ws + 98304);             // 128*128*2   = 32768 B
  float*    rpe     = (float*)(ws + 98304 + 32768);        // 4*64*64*4   = 65536 B

  prep_kernel<<<(3 * DIM * DIM + 255) / 256, 256, 0, stream>>>(
      qkv_w, proj_w, tbl, relidx, qkv_wh, proj_wh, rpe);

  win_attn_kernel<<<BWIN, 256, 0, stream>>>(
      x, mask, qkv_b, proj_b, qkv_wh, proj_wh, rpe, (float*)d_out);
}
